// SDNN_21105469293194
// MI455X (gfx1250) — matrix operand order repaired and checked by reading, not compiled, not run on hardware
//
#include <hip/hip_runtime.h>
#include <hip/hip_bf16.h>

// Problem constants (from reference)
#define B_SZ   2048   // batch
#define IN_N   1024   // IN_NUM
#define H_N    1024   // INIT_SIZE (active hidden width)
#define MAXW   2048   // MAX_SIZE
#define OUT_N  1000   // OUT_NUM
#define OUT_PAD 1024  // padded N for WMMA tiles

typedef __bf16 bf16_t;
typedef __attribute__((ext_vector_type(16))) __bf16 v16bf;
typedef __attribute__((ext_vector_type(8)))  float  v8f;
typedef __attribute__((ext_vector_type(4)))  unsigned int u32x4;
typedef __attribute__((ext_vector_type(8)))  int i32x8;
typedef __attribute__((ext_vector_type(4)))  int i32x4;

#if __has_builtin(__builtin_amdgcn_tensor_load_to_lds)
#define HAVE_TDM 1
#else
#define HAVE_TDM 0
#endif

__device__ __forceinline__ v8f wmma_bf16(v16bf a, v16bf b, v8f c) {
  return __builtin_amdgcn_wmma_f32_16x16x32_bf16(
      false, a, false, b, (short)0, c, false, false);
}

#if HAVE_TDM
// 2-D TDM descriptor per CDNA5 ISA:
//   copies `nrows` rows of `row_units` bf16 elements (data_size=2B) from
//   global (row stride `stride_units`) into contiguous LDS at lds_byte_addr.
// clang-23 / therock toolchain: 6-arg builtin form
//   (u32x4 g0, i32x8 g1, i32x4 g2, i32x4 g3, i32x8 extra, i32 cpol)
__device__ __forceinline__ void tdm_load_2d(unsigned int lds_byte_addr,
                                            const void* gptr,
                                            unsigned int row_units,
                                            unsigned int nrows,
                                            unsigned int stride_units) {
  unsigned long long ga = (unsigned long long)(uintptr_t)gptr;
  u32x4 g0;
  g0[0] = 1u;                                         // count=1 user descriptor
  g0[1] = lds_byte_addr;                              // lds_addr [63:32]
  g0[2] = (unsigned int)(ga & 0xFFFFFFFFu);           // global_addr[31:0]
  g0[3] = (unsigned int)((ga >> 32) & 0x01FFFFFFu)    // global_addr[56:32]
          | (2u << 30);                               // type=2 ("image")
  i32x8 g1;
  g1[0] = (int)(1u << 16);                            // data_size=1 (2 bytes)
  g1[1] = (int)(row_units << 16);                     // tensor_dim0[15:0]
  g1[2] = (int)((row_units >> 16) & 0xFFFFu)          // tensor_dim0[31:16]
          | (int)(nrows << 16);                       // tensor_dim1[15:0]
  g1[3] = (int)((row_units & 0xFFFFu) << 16);         // tile_dim0
  g1[4] = (int)(nrows & 0xFFFFu);                     // tile_dim1 (tile_dim2=0)
  g1[5] = (int)stride_units;                          // tensor_dim0_stride[31:0]
  g1[6] = 0;                                          // stride hi / dim1_stride
  g1[7] = 0;
  i32x4 g2 = {0, 0, 0, 0};
  i32x4 g3 = {0, 0, 0, 0};
  i32x8 gx = {0, 0, 0, 0, 0, 0, 0, 0};
  __builtin_amdgcn_tensor_load_to_lds(g0, g1, g2, g3, gx, 0);
}
#endif

// ---------------------------------------------------------------------------
// Fragment-swizzled layouts (wave32 WMMA, CDNA5 ISA §7.12.2):
//  A (16x32 bf16): lane l -> row m=l&15; kb=(l>>4)*8; e<8 -> k=kb+e else 16+kb+e-8
//  B (32x16 bf16): lane l -> col n=l&15; kb=(l>>4)*8; e<8 -> k=kb+e else 16+kb+e-8
// Flat half index in packed buffer: (tile*32 + lane)*16 + e -> 32B/lane.
// ---------------------------------------------------------------------------
__device__ __forceinline__ void a_swz_src(int f, int KT, int& m, int& k) {
  int e = f & 15, l = (f >> 4) & 31, t = f >> 9;
  int mi = t / KT, ki = t % KT;
  m = mi * 16 + (l & 15);
  int kb = (l >> 4) * 8;
  k = ki * 32 + ((e < 8) ? (kb + e) : (16 + kb + (e - 8)));
}
__device__ __forceinline__ void b_swz_src(int f, int NT, int& k, int& n) {
  int e = f & 15, l = (f >> 4) & 31, t = f >> 9;
  int ki = t / NT, ni = t % NT;
  n = ni * 16 + (l & 15);
  k = ki * 32 + ((e < 8) ? ((l >> 4) * 8 + e) : (16 + (l >> 4) * 8 + (e - 8)));
}

// ---------------- pack kernels: f32 inputs -> bf16 swizzled ----------------
__global__ void pack_x_a(const float* __restrict__ x, bf16_t* __restrict__ dst) {
  int f = blockIdx.x * blockDim.x + threadIdx.x;
  int m, k; a_swz_src(f, IN_N / 32, m, k);
  dst[f] = (bf16_t)x[m * IN_N + k];
}
__global__ void pack_w1_b(const float* __restrict__ w1, const float* __restrict__ m1,
                          bf16_t* __restrict__ dst) {
  int f = blockIdx.x * blockDim.x + threadIdx.x;
  int k, n; b_swz_src(f, H_N / 16, k, n);
  dst[f] = (bf16_t)(w1[k * MAXW + n] * m1[k * MAXW + n]);
}
__global__ void pack_w2_b(const float* __restrict__ w2, bf16_t* __restrict__ dst) {
  int f = blockIdx.x * blockDim.x + threadIdx.x;
  int k, n; b_swz_src(f, H_N / 16, k, n);
  dst[f] = (bf16_t)w2[k * MAXW + n];
}
__global__ void pack_w3_b(const float* __restrict__ w3, bf16_t* __restrict__ dst) {
  int f = blockIdx.x * blockDim.x + threadIdx.x;
  int k, n; b_swz_src(f, OUT_PAD / 16, k, n);
  dst[f] = (n < OUT_N) ? (bf16_t)w3[k * OUT_N + n] : (bf16_t)0.0f;
}
__global__ void pack_w4_b(const float* __restrict__ w4, bf16_t* __restrict__ dst) {
  int f = blockIdx.x * blockDim.x + threadIdx.x;
  int k, n; b_swz_src(f, OUT_PAD / 16, k, n);
  dst[f] = (n < OUT_N) ? (bf16_t)w4[k * OUT_N + n] : (bf16_t)0.0f;
}

// ---------------- x_proj = x @ (w1*m1)[:, :1024], 2x2 register blocking ----
__global__ void __launch_bounds__(32)
gemm_proj(const bf16_t* __restrict__ Xa, const bf16_t* __restrict__ W1b,
          float* __restrict__ xproj) {
  const int nB = blockIdx.x, mB = blockIdx.y, lane = threadIdx.x;
  const int KT = IN_N / 32, NT = H_N / 16;
  const int mi0 = 2 * mB, ni0 = 2 * nB;
  v8f c00 = {}, c01 = {}, c10 = {}, c11 = {};
  for (int ki = 0; ki < KT; ++ki) {
    const bf16_t* pa0 = Xa  + ((((mi0 * KT + ki) * 32) + lane) << 4);
    const bf16_t* pa1 = pa0 + ((KT * 32) << 4);
    const bf16_t* pb0 = W1b + ((((ki * NT + ni0) * 32) + lane) << 4);
    const bf16_t* pb1 = pb0 + (32 << 4);
    __builtin_prefetch(pa0 + (32 << 4), 0, 3);
    __builtin_prefetch(pb0 + ((NT * 32) << 4), 0, 3);
    v16bf a0 = *(const v16bf*)pa0, a1 = *(const v16bf*)pa1;
    v16bf b0 = *(const v16bf*)pb0, b1 = *(const v16bf*)pb1;
    c00 = wmma_bf16(a0, b0, c00);
    c01 = wmma_bf16(a0, b1, c01);
    c10 = wmma_bf16(a1, b0, c10);
    c11 = wmma_bf16(a1, b1, c11);
  }
  const int nn = lane & 15, rb = (lane < 16) ? 0 : 8;
#pragma unroll
  for (int r = 0; r < 8; ++r) {
    xproj[((mi0)     * 16 + rb + r) * H_N + (ni0)     * 16 + nn] = c00[r];
    xproj[((mi0)     * 16 + rb + r) * H_N + (ni0 + 1) * 16 + nn] = c01[r];
    xproj[((mi0 + 1) * 16 + rb + r) * H_N + (ni0)     * 16 + nn] = c10[r];
    xproj[((mi0 + 1) * 16 + rb + r) * H_N + (ni0 + 1) * 16 + nn] = c11[r];
  }
}

// ---------------- sequential recurrence, 16 batch rows per WG --------------
// h_j = relu(x_proj[:,j] + sum_{k<j} h_k*w2[k,j] + b1[j]), blocked by 32 j's.
// Cross-block term via WMMA against TDM-staged w2 fragment panels in LDS.
#define CHUNK 8   // ki rows per TDM panel chunk (8 * 2KB = 16KB LDS)
__global__ void __launch_bounds__(32)
recurrence(const float* __restrict__ xproj, const float* __restrict__ w2,
           const float* __restrict__ b1, const bf16_t* __restrict__ W2b,
           bf16_t* __restrict__ Ha) {
  __shared__ bf16_t hrow[16][H_N];          // 32 KB hidden slab
  __shared__ bf16_t Bpan[CHUNK * 1024];     // 16 KB TDM panel buffer
  __shared__ float  accs[16][32];
  __shared__ float  hs[16][32];
  __shared__ float  w2blk[32][32];
  const int lane = threadIdx.x;
  const int mi = blockIdx.x, m0 = mi * 16;
  const unsigned int pan_lds =
      (unsigned int)(uintptr_t)(void*)&Bpan[0];   // LDS aperture: addr[31:0]

  for (int bi = 0; bi < 32; ++bi) {
    const int jb = bi * 32;
    for (int idx = lane; idx < 32 * 32; idx += 32) {
      int k = idx >> 5, jj = idx & 31;
      w2blk[k][jj] = w2[(jb + k) * MAXW + (jb + jj)];
    }
    v8f c0 = {}, c1 = {};
    const int m = lane & 15, kb = (lane >> 4) * 8;
    // K = 32*bi, processed in CHUNK-row panels of packed B fragments.
    for (int cc = 0; cc * CHUNK < bi; ++cc) {
      const int rows = (bi - cc * CHUNK < CHUNK) ? (bi - cc * CHUNK) : CHUNK;
      const bf16_t* gsrc = W2b + (size_t)(cc * CHUNK) * 32768 + (size_t)bi * 1024;
#if HAVE_TDM
      asm volatile("s_wait_dscnt 0x0" ::: "memory");   // prior ds reads drained
      tdm_load_2d(pan_lds, gsrc, 1024u, (unsigned int)rows, 32768u);
      __builtin_amdgcn_s_wait_tensorcnt(0);
#else
      __syncthreads();
      for (int idx = lane; idx < rows * 128; idx += 32) {
        int r = idx >> 7, col = idx & 127;
        ((uint4*)Bpan)[r * 128 + col] =
            ((const uint4*)(gsrc + (size_t)r * 32768))[col];
      }
#endif
      __syncthreads();
      for (int r = 0; r < rows; ++r) {
        const int k0 = (cc * CHUNK + r) * 32;
        v16bf a;
        const bf16_t* p0 = &hrow[m][k0 + kb];
        const bf16_t* p1 = &hrow[m][k0 + 16 + kb];
#pragma unroll
        for (int e = 0; e < 8; ++e) { a[e] = p0[e]; a[8 + e] = p1[e]; }
        v16bf b0  = *(const v16bf*)(Bpan + r * 1024 + lane * 16);
        v16bf b1v = *(const v16bf*)(Bpan + r * 1024 + 512 + lane * 16);
        c0 = wmma_bf16(a, b0, c0);
        c1 = wmma_bf16(a, b1v, c1);
      }
    }
    __syncthreads();
    {
      const int n = lane & 15, rb = (lane < 16) ? 0 : 8;
#pragma unroll
      for (int r = 0; r < 8; ++r) {
        accs[rb + r][n]      = c0[r];
        accs[rb + r][16 + n] = c1[r];
      }
    }
    __syncthreads();
    if (lane < 16) {                 // triangular tail, lanes 0..15 = rows
      const int mr = lane;
      for (int jj = 0; jj < 32; ++jj) {
        const int j = jb + jj;
        float val = accs[mr][jj] + xproj[(m0 + mr) * H_N + j] + b1[j];
        for (int k = 0; k < jj; ++k) val += hs[mr][k] * w2blk[k][jj];
        val = fmaxf(val, 0.0f);
        hs[mr][jj] = val;
        hrow[mr][j] = (bf16_t)val;
      }
    }
    __syncthreads();
  }

  // emit hidden slab in A-swizzled bf16 layout for the output GEMM
  const int m = lane & 15, kb = (lane >> 4) * 8;
  for (int ki = 0; ki < 32; ++ki) {
    v16bf a;
    const bf16_t* p0 = &hrow[m][ki * 32 + kb];
    const bf16_t* p1 = &hrow[m][ki * 32 + 16 + kb];
#pragma unroll
    for (int e = 0; e < 8; ++e) { a[e] = p0[e]; a[8 + e] = p1[e]; }
    *(v16bf*)(Ha + ((((mi * 32 + ki) * 32) + lane) << 4)) = a;
  }
}

// ---------------- out = hidden @ w3 + x @ w4 + b2, 2x2 blocking ------------
__global__ void __launch_bounds__(32)
gemm_out(const bf16_t* __restrict__ Ha, const bf16_t* __restrict__ Xa,
         const bf16_t* __restrict__ W3b, const bf16_t* __restrict__ W4b,
         const float* __restrict__ b2, float* __restrict__ out) {
  const int nB = blockIdx.x, mB = blockIdx.y, lane = threadIdx.x;
  const int KT = H_N / 32, NT = OUT_PAD / 16;
  const int mi0 = 2 * mB, ni0 = 2 * nB;
  v8f c00 = {}, c01 = {}, c10 = {}, c11 = {};
  for (int ki = 0; ki < KT; ++ki) {
    const bf16_t* pa0 = Ha  + ((((mi0 * KT + ki) * 32) + lane) << 4);
    const bf16_t* pa1 = pa0 + ((KT * 32) << 4);
    const bf16_t* pb0 = W3b + ((((ki * NT + ni0) * 32) + lane) << 4);
    const bf16_t* pb1 = pb0 + (32 << 4);
    v16bf a0 = *(const v16bf*)pa0, a1 = *(const v16bf*)pa1;
    v16bf b0 = *(const v16bf*)pb0, b1 = *(const v16bf*)pb1;
    c00 = wmma_bf16(a0, b0, c00);
    c01 = wmma_bf16(a0, b1, c01);
    c10 = wmma_bf16(a1, b0, c10);
    c11 = wmma_bf16(a1, b1, c11);
  }
  for (int ki = 0; ki < KT; ++ki) {
    const bf16_t* pa0 = Xa  + ((((mi0 * KT + ki) * 32) + lane) << 4);
    const bf16_t* pa1 = pa0 + ((KT * 32) << 4);
    const bf16_t* pb0 = W4b + ((((ki * NT + ni0) * 32) + lane) << 4);
    const bf16_t* pb1 = pb0 + (32 << 4);
    v16bf a0 = *(const v16bf*)pa0, a1 = *(const v16bf*)pa1;
    v16bf b0 = *(const v16bf*)pb0, b1 = *(const v16bf*)pb1;
    c00 = wmma_bf16(a0, b0, c00);
    c01 = wmma_bf16(a0, b1, c01);
    c10 = wmma_bf16(a1, b0, c10);
    c11 = wmma_bf16(a1, b1, c11);
  }
  const int nn = lane & 15, rb = (lane < 16) ? 0 : 8;
  const int n0 = ni0 * 16 + nn, n1 = (ni0 + 1) * 16 + nn;
  const float bias0 = (n0 < OUT_N) ? b2[n0] : 0.0f;
  const float bias1 = (n1 < OUT_N) ? b2[n1] : 0.0f;
#pragma unroll
  for (int r = 0; r < 8; ++r) {
    int ra = mi0 * 16 + rb + r, rc = (mi0 + 1) * 16 + rb + r;
    if (n0 < OUT_N) { out[ra * OUT_N + n0] = c00[r] + bias0;
                      out[rc * OUT_N + n0] = c10[r] + bias0; }
    if (n1 < OUT_N) { out[ra * OUT_N + n1] = c01[r] + bias1;
                      out[rc * OUT_N + n1] = c11[r] + bias1; }
  }
}

// ---------------------------------------------------------------------------
extern "C" void kernel_launch(void* const* d_in, const int* in_sizes, int n_in,
                              void* d_out, int out_size, void* d_ws, size_t ws_size,
                              hipStream_t stream) {
  (void)in_sizes; (void)n_in; (void)out_size; (void)ws_size;
  const float* x  = (const float*)d_in[0];
  const float* w1 = (const float*)d_in[1];
  const float* w2 = (const float*)d_in[2];
  const float* w3 = (const float*)d_in[3];
  const float* w4 = (const float*)d_in[4];
  const float* b1 = (const float*)d_in[5];
  const float* b2 = (const float*)d_in[6];
  const float* m1 = (const float*)d_in[7];

  char* ws = (char*)d_ws;                              // ~24 MB used
  bf16_t* Xa  = (bf16_t*)(ws + (size_t)0);             // 4 MB  x, A-swizzled
  bf16_t* Ha  = (bf16_t*)(ws + ((size_t)4  << 20));    // 4 MB  hidden, A-swizzled
  bf16_t* W1b = (bf16_t*)(ws + ((size_t)8  << 20));    // 2 MB
  bf16_t* W2b = (bf16_t*)(ws + ((size_t)10 << 20));    // 2 MB
  bf16_t* W3b = (bf16_t*)(ws + ((size_t)12 << 20));    // 2 MB
  bf16_t* W4b = (bf16_t*)(ws + ((size_t)14 << 20));    // 2 MB
  float*  XP  = (float*) (ws + ((size_t)16 << 20));    // 8 MB  x_proj f32

  pack_x_a <<<(B_SZ * IN_N)    / 256, 256, 0, stream>>>(x, Xa);
  pack_w1_b<<<(IN_N * H_N)     / 256, 256, 0, stream>>>(w1, m1, W1b);
  pack_w2_b<<<(H_N * H_N)      / 256, 256, 0, stream>>>(w2, W2b);
  pack_w3_b<<<(H_N * OUT_PAD)  / 256, 256, 0, stream>>>(w3, W3b);
  pack_w4_b<<<(IN_N * OUT_PAD) / 256, 256, 0, stream>>>(w4, W4b);

  gemm_proj<<<dim3(H_N / 32, B_SZ / 32), 32, 0, stream>>>(Xa, W1b, XP);
  recurrence<<<B_SZ / 16, 32, 0, stream>>>(XP, w2, b1, W2b, Ha);
  gemm_out<<<dim3(OUT_PAD / 32, B_SZ / 32), 32, 0, stream>>>(
      Ha, Xa, W3b, W4b, b2, (float*)d_out);
}
